// PairWeightedAveraging_515396076367
// MI455X (gfx1250) — matrix operand order repaired and checked by reading, not compiled, not run on hardware
//
#include <hip/hip_runtime.h>
#include <cstdint>

// Pair-weighted averaging (AlphaFold MSA column-wise attention style).
// S=256 seqs, N=512 tokens, Dm=64, Dz=128, H=8 heads, HD=32 head dim.
#define S_  256
#define N_  512
#define DM_ 64
#define DZ_ 128
#define H_  8
#define HD_ 32
#define HK_ 256  // H_*HD_

typedef __attribute__((ext_vector_type(16))) __bf16 v16bf;
typedef __attribute__((ext_vector_type(8)))  float  v8f;

union BF16Frag {
  v16bf v;
  unsigned short us[16];
  uint4 q[2];
};
union Pack8 {
  unsigned short us[8];
  uint4 q;
};

__device__ __forceinline__ unsigned short f2bf(float f) {
  unsigned int u = __float_as_uint(f);
  u += 0x7FFFu + ((u >> 16) & 1u);          // round-to-nearest-even
  return (unsigned short)(u >> 16);
}
__device__ __forceinline__ float bf2f(unsigned short h) {
  return __uint_as_float(((unsigned int)h) << 16);
}

// ---------------------------------------------------------------------------
// K0: pre-pack an fp32 row-major weight [K x Nc] into WMMA B-fragment order:
//     frag(kblk,nblk) = 32 lanes x 16 bf16; lane l holds B[k=32*kblk+16*(l>>4)+idx][n]
//     with n = 16*nblk + (l&15).  One 32-thread block per fragment.
// ---------------------------------------------------------------------------
__global__ __launch_bounds__(32) void pwa_pack_b(
    const float* __restrict__ src, unsigned short* __restrict__ dst, int Nc) {
  const int lane = threadIdx.x;
  const int nblks = Nc >> 4;
  const int kblk = blockIdx.x / nblks, nblk = blockIdx.x % nblks;
  const int n  = nblk * 16 + (lane & 15);
  const int kh = kblk * 32 + 16 * (lane >> 4);
  BF16Frag b;
  for (int idx = 0; idx < 16; ++idx)
    b.us[idx] = f2bf(src[(kh + idx) * Nc + n]);
  uint4* d = (uint4*)(dst + ((long)blockIdx.x * 32 + lane) * 16);
  d[0] = b.q[0];
  d[1] = b.q[1];
}

// ---------------------------------------------------------------------------
// K1: layernorm(z) + b[h,i,j] = zn(i,j,:) . Wz[:,h]   (one wave per (i,j) row)
// ---------------------------------------------------------------------------
__global__ __launch_bounds__(256) void pwa_znorm_bias(
    const float* __restrict__ z, const float* __restrict__ zw,
    const float* __restrict__ zb, const float* __restrict__ Wz,
    float* __restrict__ b) {
  const int wave = threadIdx.x >> 5, lane = threadIdx.x & 31;
  const long row = (long)blockIdx.x * 8 + wave;           // i*N + j
  const float4 xv = *(const float4*)(z + row * DZ_ + lane * 4);
  float x[4] = {xv.x, xv.y, xv.z, xv.w};
  float ssum = x[0] + x[1] + x[2] + x[3];
  for (int msk = 16; msk >= 1; msk >>= 1) ssum += __shfl_xor(ssum, msk, 32);
  const float mu = ssum * (1.f / DZ_);
  float vs = 0.f;
  for (int t = 0; t < 4; ++t) { float d = x[t] - mu; vs += d * d; }
  for (int msk = 16; msk >= 1; msk >>= 1) vs += __shfl_xor(vs, msk, 32);
  const float rstd = rsqrtf(vs * (1.f / DZ_) + 1e-5f);
  float xn[4];
  for (int t = 0; t < 4; ++t) {
    int cc = lane * 4 + t;
    xn[t] = (x[t] - mu) * rstd * zw[cc] + zb[cc];
  }
  for (int h = 0; h < H_; ++h) {
    float p = 0.f;
    for (int t = 0; t < 4; ++t) { int cc = lane * 4 + t; p += xn[t] * Wz[cc * H_ + h]; }
    for (int msk = 16; msk >= 1; msk >>= 1) p += __shfl_xor(p, msk, 32);
    if (lane == h) b[(long)h * N_ * N_ + row] = p;
  }
}

// ---------------------------------------------------------------------------
// K2: softmax over j for each (h,i) row -> w bf16 (row-major; A-operand friendly)
// ---------------------------------------------------------------------------
__global__ __launch_bounds__(256) void pwa_softmax(
    const float* __restrict__ b, unsigned short* __restrict__ wbf) {
  __shared__ float red[256];
  const long row = blockIdx.x;                 // h*N + i
  const float* br = b + row * N_;
  const int t = threadIdx.x;
  const float a0 = br[t], a1 = br[t + 256];
  red[t] = fmaxf(a0, a1);
  __syncthreads();
  for (int s = 128; s >= 1; s >>= 1) { if (t < s) red[t] = fmaxf(red[t], red[t + s]); __syncthreads(); }
  const float mx = red[0];
  __syncthreads();
  const float e0 = __expf(a0 - mx), e1 = __expf(a1 - mx);
  red[t] = e0 + e1;
  __syncthreads();
  for (int s = 128; s >= 1; s >>= 1) { if (t < s) red[t] += red[t + s]; __syncthreads(); }
  const float inv = __builtin_amdgcn_rcpf(red[0]);
  wbf[row * N_ + t]       = f2bf(e0 * inv);
  wbf[row * N_ + t + 256] = f2bf(e1 * inv);
}

// ---------------------------------------------------------------------------
// K3: layernorm(m) on a 16-row (s,j) tile, then WMMA projections.
//   Value tiles  -> vfrag: pre-swizzled B-fragment layout [h][jblk][nblk][lane][16]
//                   (each lane's 8 D elements are contiguous -> one uint4 store)
//   Gate tiles   -> g stored [h][n=s*32+k][j] so K4 reads one uint4 per lane.
//   Two separate loops keep the value/gate code paths branch-free.
// ---------------------------------------------------------------------------
__global__ __launch_bounds__(256) void pwa_mnorm_proj(
    const float* __restrict__ m, const float* __restrict__ mw, const float* __restrict__ mb,
    const unsigned short* __restrict__ wmfrag, const unsigned short* __restrict__ wgfrag,
    unsigned short* __restrict__ vfrag, unsigned short* __restrict__ gbuf) {
  __shared__ unsigned short mn[16 * DM_];      // 16 normalized rows, bf16
  const int tid = threadIdx.x;
  const int wave = tid >> 5, lane = tid & 31;
  const long r0 = (long)blockIdx.x * 16;       // flattened (s, j) row base
  const int s  = (int)(r0 >> 9);               // 16 | 512 => same s for whole tile
  const int j0 = (int)(r0 & (N_ - 1));

  {  // wave handles rows 2w and 2w+1 via lane halves (16 lanes x 4 elems = 64)
    const int lrn = 2 * wave + (lane >> 4);
    const int gl  = lane & 15;
    const float4 xv = *(const float4*)(m + (r0 + lrn) * DM_ + gl * 4);
    float x[4] = {xv.x, xv.y, xv.z, xv.w};
    float ssum = x[0] + x[1] + x[2] + x[3];
    for (int msk = 8; msk >= 1; msk >>= 1) ssum += __shfl_xor(ssum, msk, 32);
    const float mu = ssum * (1.f / DM_);
    float vs = 0.f;
    for (int t = 0; t < 4; ++t) { float d = x[t] - mu; vs += d * d; }
    for (int msk = 8; msk >= 1; msk >>= 1) vs += __shfl_xor(vs, msk, 32);
    const float rstd = rsqrtf(vs * (1.f / DM_) + 1e-5f);
    for (int t = 0; t < 4; ++t) {
      int cc = gl * 4 + t;
      mn[lrn * DM_ + cc] = f2bf((x[t] - mu) * rstd * mw[cc] + mb[cc]);
    }
  }
  __syncthreads();

  // A fragments (16x32 bf16, K=0..31 and K=32..63) shared by all tiles of this wave
  const int lr = lane & 15;
  const int base8 = (lane >> 4) * 16;
  BF16Frag a0, a1;
  a0.q[0] = *(const uint4*)(mn + lr * DM_ +  0 + base8);
  a0.q[1] = *(const uint4*)(mn + lr * DM_ +  8 + base8);
  a1.q[0] = *(const uint4*)(mn + lr * DM_ + 32 + base8);
  a1.q[1] = *(const uint4*)(mn + lr * DM_ + 40 + base8);

  const int jblk = j0 >> 5;
  const int dstlane = ((j0 >> 4) & 1) * 16 + (lane & 15);

  // ---- value tiles: ct = wave*2 + t (16 tiles of 16 cols over HK=256) ----
  for (int t = 0; t < 2; ++t) {
    const int ct = wave * 2 + t;
    BF16Frag b0, b1;                           // Wm frags: kblk 0 and 1
    const unsigned short* f0 = wmfrag + ((long)(0 * 16 + ct) * 32 + lane) * 16;
    const unsigned short* f1 = wmfrag + ((long)(1 * 16 + ct) * 32 + lane) * 16;
    b0.q[0] = *(const uint4*)(f0); b0.q[1] = *(const uint4*)(f0 + 8);
    b1.q[0] = *(const uint4*)(f1); b1.q[1] = *(const uint4*)(f1 + 8);
    v8f c = {};
    c = __builtin_amdgcn_wmma_f32_16x16x32_bf16(false, a0.v, false, b0.v, (short)0, c, false, false);
    c = __builtin_amdgcn_wmma_f32_16x16x32_bf16(false, a1.v, false, b1.v, (short)0, c, false, false);
    const int hh = ct >> 1;
    const int nblk = s * 2 + (ct & 1);
    Pack8 pk;
    for (int ii = 0; ii < 8; ++ii) pk.us[ii] = f2bf(c[ii]);
    *(uint4*)(vfrag + ((((long)hh * 16 + jblk) * 512 + nblk) * 32 + dstlane) * 16
                    + 8 * (lane >> 4)) = pk.q;
  }

  // ---- gate tiles: sigmoid fused; g stored [h][n][j] ----
  for (int t = 0; t < 2; ++t) {
    const int ct = wave * 2 + t;
    BF16Frag b0, b1;                           // Wg frags
    const unsigned short* f0 = wgfrag + ((long)(0 * 16 + ct) * 32 + lane) * 16;
    const unsigned short* f1 = wgfrag + ((long)(1 * 16 + ct) * 32 + lane) * 16;
    b0.q[0] = *(const uint4*)(f0); b0.q[1] = *(const uint4*)(f0 + 8);
    b1.q[0] = *(const uint4*)(f1); b1.q[1] = *(const uint4*)(f1 + 8);
    v8f c = {};
    c = __builtin_amdgcn_wmma_f32_16x16x32_bf16(false, a0.v, false, b0.v, (short)0, c, false, false);
    c = __builtin_amdgcn_wmma_f32_16x16x32_bf16(false, a1.v, false, b1.v, (short)0, c, false, false);
    const int hh = ct >> 1;
    const int kk = (ct & 1) * 16 + (lane & 15);          // k within head
    Pack8 pk;
    for (int ii = 0; ii < 8; ++ii)
      pk.us[ii] = f2bf(__builtin_amdgcn_rcpf(1.f + __expf(-c[ii])));
    *(uint4*)(gbuf + ((long)(hh * 8192) + s * 32 + kk) * 512 + j0 + 8 * (lane >> 4)) = pk.q;
  }
}

// ---------------------------------------------------------------------------
// K4: per head h: O = W_h[512 x 512] x V_h[512 x 8192], gate, emit og bf16.
//     Each block: 64 i-rows x 128 cols.  Each wave: ONE B stream shared by
//     FOUR independent accumulator chains (4 i-tiles) -> 4x ILP on the XDL
//     pipe and 4x less vfrag (B) re-read traffic.  No LDS, no barriers.
// ---------------------------------------------------------------------------
__global__ __launch_bounds__(256) void pwa_attn(
    const unsigned short* __restrict__ wbf, const unsigned short* __restrict__ vfrag,
    const unsigned short* __restrict__ gbuf, unsigned short* __restrict__ ogbf) {
  const int tid = threadIdx.x;
  const int wave = tid >> 5, lane = tid & 31;
  int bid = blockIdx.x;
  const int cb = bid & 63;  bid >>= 6;       // 64 col blocks of 128
  const int it = bid & 7;   bid >>= 3;       // 8 row blocks of 64
  const int h  = bid;                        // 8 heads
  const int i0 = it * 64;
  const int colw = cb * 128 + wave * 16;

  const int lr = lane & 15;
  const int base8 = (lane >> 4) * 16;
  const int nblk = colw >> 4;

  v8f acc[4];
  for (int tt = 0; tt < 4; ++tt) acc[tt] = (v8f){};
  const unsigned short* arow = wbf + ((long)(h * N_ + i0 + lr)) * N_ + base8;
  const unsigned short* bcol = vfrag + (((long)h * 16 * 512 + nblk) * 32 + lane) * 16;

  for (int kb = 0; kb < N_; kb += 32) {
    BF16Frag b;
    const unsigned short* br = bcol + (long)(kb >> 5) * 512 * 32 * 16;
    b.q[0] = *(const uint4*)(br);
    b.q[1] = *(const uint4*)(br + 8);
#pragma unroll
    for (int tt = 0; tt < 4; ++tt) {
      BF16Frag a;
      const unsigned short* ar = arow + (long)tt * 16 * N_ + kb;
      a.q[0] = *(const uint4*)(ar);
      a.q[1] = *(const uint4*)(ar + 8);
      acc[tt] = __builtin_amdgcn_wmma_f32_16x16x32_bf16(
          false, a.v, false, b.v, (short)0, acc[tt], false, false);
    }
  }

  const int n = colw + (lane & 15);
  const int s = n >> 5, k = n & 31;
  const unsigned short* gp = gbuf + ((long)(h * 8192) + n) * 512 + i0 + 8 * (lane >> 4);
#pragma unroll
  for (int tt = 0; tt < 4; ++tt) {
    Pack8 gq;
    gq.q = *(const uint4*)(gp + tt * 16);
    for (int ii = 0; ii < 8; ++ii) {
      const int gi = i0 + tt * 16 + 8 * (lane >> 4) + ii;
      ogbf[((long)(s * N_ + gi)) * HK_ + h * HD_ + k] = f2bf(acc[tt][ii] * bf2f(gq.us[ii]));
    }
  }
}

// ---------------------------------------------------------------------------
// K5: out[(s,i), c] = og[(s,i), hk] x Wo[hk, c]   (M=131072, K=256, N=64)
//     Two independent accumulator chains over even/odd k-blocks (2x ILP).
// ---------------------------------------------------------------------------
__global__ __launch_bounds__(256) void pwa_outproj(
    const unsigned short* __restrict__ ogbf, const unsigned short* __restrict__ wofrag,
    float* __restrict__ out) {
  const int tid = threadIdx.x;
  const int wave = tid >> 5, lane = tid & 31;
  const int rt = wave >> 2, ct = wave & 3;     // 2 row tiles x 4 col tiles
  const long r0 = (long)blockIdx.x * 32 + rt * 16;
  const int lr = lane & 15;
  const int base8 = (lane >> 4) * 16;
  v8f c0 = {}, c1 = {};
  for (int kb = 0; kb < HK_; kb += 64) {
    BF16Frag a0, a1, b0, b1;
    const unsigned short* ar = ogbf + (r0 + lr) * HK_ + kb + base8;
    a0.q[0] = *(const uint4*)(ar);
    a0.q[1] = *(const uint4*)(ar + 8);
    a1.q[0] = *(const uint4*)(ar + 32);
    a1.q[1] = *(const uint4*)(ar + 40);
    const unsigned short* br = wofrag + ((long)((kb >> 5) * 4 + ct) * 32 + lane) * 16;
    b0.q[0] = *(const uint4*)(br);
    b0.q[1] = *(const uint4*)(br + 8);
    b1.q[0] = *(const uint4*)(br + 4 * 32 * 16);
    b1.q[1] = *(const uint4*)(br + 4 * 32 * 16 + 8);
    c0 = __builtin_amdgcn_wmma_f32_16x16x32_bf16(false, a0.v, false, b0.v, (short)0, c0, false, false);
    c1 = __builtin_amdgcn_wmma_f32_16x16x32_bf16(false, a1.v, false, b1.v, (short)0, c1, false, false);
  }
  for (int ii = 0; ii < 8; ++ii) {
    const long row = r0 + ii + 8 * (lane >> 4);
    out[row * DM_ + ct * 16 + (lane & 15)] = c0[ii] + c1[ii];
  }
}

// ---------------------------------------------------------------------------
extern "C" void kernel_launch(void* const* d_in, const int* in_sizes, int n_in,
                              void* d_out, int out_size, void* d_ws, size_t ws_size,
                              hipStream_t stream) {
  (void)in_sizes; (void)n_in; (void)out_size; (void)ws_size;
  const float* m  = (const float*)d_in[0];
  const float* z  = (const float*)d_in[1];
  const float* mw = (const float*)d_in[2];
  const float* mb = (const float*)d_in[3];
  const float* zw = (const float*)d_in[4];
  const float* zb = (const float*)d_in[5];
  const float* Wm = (const float*)d_in[6];
  const float* Wg = (const float*)d_in[7];
  const float* Wz = (const float*)d_in[8];
  const float* Wo = (const float*)d_in[9];
  float* out = (float*)d_out;

  // Workspace layout (~204 MB):
  //   b      fp32 [H][N][N]              8 MB
  //   w      bf16 [H][N][N]              4 MB
  //   vfrag  bf16 frag [h][jblk][nblk]  64 MB
  //   g      bf16 [h][n][j]             64 MB
  //   og     bf16 [S*N][HK]             64 MB
  //   wm/wg/wo frag bf16                32 KB each
  char* ws = (char*)d_ws;
  float*          b      = (float*)ws;
  unsigned short* wbf    = (unsigned short*)(ws + (size_t)H_ * N_ * N_ * 4);
  unsigned short* vfrag  = (unsigned short*)((char*)wbf   + (size_t)H_ * N_ * N_ * 2);
  unsigned short* gbuf   = (unsigned short*)((char*)vfrag + (size_t)H_ * N_ * S_ * HD_ * 2);
  unsigned short* ogbf   = (unsigned short*)((char*)gbuf  + (size_t)S_ * N_ * HK_ * 2);
  unsigned short* wmfrag = (unsigned short*)((char*)ogbf  + (size_t)S_ * N_ * HK_ * 2);
  unsigned short* wgfrag = wmfrag + 32 * 32 * 16;   // 32 frags of 512 bf16
  unsigned short* wofrag = wgfrag + 32 * 32 * 16;

  pwa_pack_b<<<(DM_ / 32) * (HK_ / 16), 32, 0, stream>>>(Wm, wmfrag, HK_);  // 32 frags
  pwa_pack_b<<<(DM_ / 32) * (HK_ / 16), 32, 0, stream>>>(Wg, wgfrag, HK_);  // 32 frags
  pwa_pack_b<<<(HK_ / 32) * (DM_ / 16), 32, 0, stream>>>(Wo, wofrag, DM_);  // 32 frags

  pwa_znorm_bias<<<N_ * N_ / 8,  256, 0, stream>>>(z, zw, zb, Wz, b);
  pwa_softmax   <<<H_ * N_,      256, 0, stream>>>(b, wbf);
  pwa_mnorm_proj<<<S_ * N_ / 16, 256, 0, stream>>>(m, mw, mb, wmfrag, wgfrag, vfrag, gbuf);
  pwa_attn      <<<H_ * 8 * 64,  256, 0, stream>>>(wbf, vfrag, gbuf, ogbf);
  pwa_outproj   <<<S_ * N_ / 32, 256, 0, stream>>>(ogbf, wofrag, out);
}
